// MultiHeadedAttention_13108240187887
// MI455X (gfx1250) — matrix operand order repaired and checked by reading, not compiled, not run on hardware
//
#include <hip/hip_runtime.h>

// MultiHeadedAttention (MAT-style graph attention) for MI455X / gfx1250.
// All GEMMs on v_wmma_f32_16x16x32_f16 (f16 operands, f32 accumulate).
// GEMM tiles streamed into LDS by the Tensor Data Mover with a double-buffered
// pipeline (tensor_load_to_lds + s_wait_tensorcnt 2/0); manual staging fallback.
// Workspace: ~61MB.

#define B_   16
#define N_   256
#define D_   1024
#define H_   16
#define DK_  64
#define LAM_ATT  0.33f
#define LAM_DIST 0.33f
#define LAM_ADJ  (1.0f - 0.33f - 0.33f)

typedef __attribute__((ext_vector_type(16))) _Float16 v16h;
typedef __attribute__((ext_vector_type(8)))  _Float16 v8h;
typedef __attribute__((ext_vector_type(4)))  _Float16 v4h;
typedef __attribute__((ext_vector_type(8)))  float    v8f;
typedef __attribute__((ext_vector_type(4)))  unsigned int u32x4;
typedef __attribute__((ext_vector_type(8)))  int      i32x8;
typedef __attribute__((ext_vector_type(4)))  int      i32x4;

#if defined(__gfx1250__) && __has_builtin(__builtin_amdgcn_tensor_load_to_lds)
#define USE_TDM 1
#else
#define USE_TDM 0
#endif

__device__ __forceinline__ v16h cat16(v8h lo, v8h hi) {
  return __builtin_shufflevector(lo, hi, 0,1,2,3,4,5,6,7,8,9,10,11,12,13,14,15);
}
__device__ __forceinline__ v8f wmma_f16(v16h a, v16h b, v8f c) {
  return __builtin_amdgcn_wmma_f32_16x16x32_f16(false, a, false, b, (short)0, c,
                                                false, false);
}

#if USE_TDM
// 2D tile load via Tensor Data Mover (CDNA5 ISA).
// Loads tile_d1 rows x tile_d0 f16-elements from global (row stride d0_stride
// elements) into LDS at lds_addr, inserting 4 DWORDs of padding every
// 16 DWORDs (i.e., after each 32-half row) -> LDS row pitch = 40 halves.
__device__ __forceinline__ void tdm_load_tile_2d(
    unsigned lds_addr, const void* gptr, unsigned long long d0_stride,
    unsigned tile_d0, unsigned tile_d1)
{
  unsigned long long ga = (unsigned long long)(uintptr_t)gptr;
  const unsigned td0 = 1u << 20, td1 = 1u << 20;   // generous tensor dims
  u32x4 g0 = { 1u,                                  // count=1 (valid)
               lds_addr,
               (unsigned)ga,
               (unsigned)((ga >> 32) & 0x01ffffffull) | 0x80000000u }; // type=2
  i32x8 g1;
  g1[0] = (int)((1u << 16)     // data_size = 1 -> 2 bytes
              | (1u << 20)     // pad_enable
              | (3u << 22)     // pad_interval code 3 = 16 DWORDs
              | (3u << 25));   // pad_amount   code 3 = 4 DWORDs
  g1[1] = (int)((td0 & 0xffffu) << 16);                              // dim0 lo
  g1[2] = (int)(((td0 >> 16) & 0xffffu) | ((td1 & 0xffffu) << 16));  // dim0 hi/dim1 lo
  g1[3] = (int)(((td1 >> 16) & 0xffffu) | (tile_d0 << 16));          // dim1 hi/tile0
  g1[4] = (int)(tile_d1 & 0xffffu);                                  // tile1, tile2=0
  g1[5] = (int)(unsigned)(d0_stride & 0xffffffffull);                // dim0 stride lo
  g1[6] = (int)(unsigned)((d0_stride >> 32) & 0xffffull);            // stride hi
  g1[7] = 0;
  i32x4 z4 = {0, 0, 0, 0};
  i32x8 z8 = {0, 0, 0, 0, 0, 0, 0, 0};
  // 6-arg form (probe-verified on this toolchain):
  // (u32x4 g0, i32x8 g1, i32x4 g2, i32x4 g3, i32x8, i32 cpol)
  __builtin_amdgcn_tensor_load_to_lds(g0, g1, z4, z4, z8, 0);
}
#endif

// ---------------------------------------------------------------------------
// f32 -> f16 elementwise (4 elems/thread).
// ---------------------------------------------------------------------------
__global__ __launch_bounds__(256) void cvt_f32_f16_k(
    const float* __restrict__ x, _Float16* __restrict__ y)
{
  size_t i = ((size_t)blockIdx.x * 256 + threadIdx.x) * 4;
  float4 f = *(const float4*)(x + i);
  v4h h = { (_Float16)f.x, (_Float16)f.y, (_Float16)f.z, (_Float16)f.w };
  *(v4h*)(y + i) = h;
}

// ---------------------------------------------------------------------------
// W[1024 k][1024 n] f32 -> WT[1024 n][1024 k] f16 (32x32 LDS tile transpose).
// ---------------------------------------------------------------------------
__global__ __launch_bounds__(256) void cvt_w_tr_k(
    const float* __restrict__ Wm, _Float16* __restrict__ WT)
{
  __shared__ float t[32][33];
  const int tx = threadIdx.x & 31, ty = threadIdx.x >> 5;   // ty 0..7
  const int k0 = blockIdx.x * 32, n0 = blockIdx.y * 32;
  #pragma unroll
  for (int i = 0; i < 4; ++i)
    t[ty + 8 * i][tx] = Wm[(size_t)(k0 + ty + 8 * i) * 1024 + n0 + tx];
  __syncthreads();
  #pragma unroll
  for (int i = 0; i < 4; ++i)
    WT[(size_t)(n0 + ty + 8 * i) * 1024 + k0 + tx] = (_Float16)t[tx][ty + 8 * i];
}

// ---------------------------------------------------------------------------
// pmix[b][i][j] = LAM_DIST*softmax_j(-dist, masked->-inf) + LAM_ADJ*rowNorm(adj)
// ---------------------------------------------------------------------------
__global__ __launch_bounds__(256) void pmix_k(
    const float* __restrict__ adj, const float* __restrict__ dist,
    const int* __restrict__ mask, _Float16* __restrict__ pmix)
{
  const int w = threadIdx.x >> 5, l = threadIdx.x & 31;
  const int row = blockIdx.x * 8 + w;      // b*256 + i
  const int b = row >> 8;
  const float* ar = adj  + (size_t)row * N_;
  const float* dr = dist + (size_t)row * N_;

  float a[8], dv[8];
  int mk[8];
  float asum = 0.f, dmax = -3.0e38f;
  #pragma unroll
  for (int t = 0; t < 8; ++t) {
    int j = l + 32 * t;
    a[t]  = ar[j];
    dv[t] = dr[j];
    mk[t] = mask[b * N_ + j];
    asum += a[t];
    if (mk[t]) dmax = fmaxf(dmax, -dv[t]);
  }
  #pragma unroll
  for (int off = 16; off >= 1; off >>= 1) {
    asum += __shfl_xor(asum, off, 32);
    dmax  = fmaxf(dmax, __shfl_xor(dmax, off, 32));
  }
  float e[8], dsum = 0.f;
  #pragma unroll
  for (int t = 0; t < 8; ++t) {
    e[t] = mk[t] ? __expf(-dv[t] - dmax) : 0.f;
    dsum += e[t];
  }
  #pragma unroll
  for (int off = 16; off >= 1; off >>= 1) dsum += __shfl_xor(dsum, off, 32);

  const float ia = LAM_ADJ  / (asum + 1e-6f);
  const float id = LAM_DIST / dsum;
  #pragma unroll
  for (int t = 0; t < 8; ++t)
    pmix[(size_t)row * N_ + l + 32 * t] = (_Float16)(a[t] * ia + e[t] * id);
}

// ---------------------------------------------------------------------------
// Unified GEMM: out = A(f16)[4096x1024] @ BT(f16)[1024n x 1024k]^T + bias.
// Block tile 128x128, 8 waves (4x2), each 2x4 WMMA tiles, K-chunks of 32.
// Double-buffered TDM pipeline: while all 8 waves run WMMAs on buffer b, the
// Tensor Data Mover streams chunk k+1 into buffer b^1. TENSORcnt completes
// in order, so s_wait_tensorcnt(2) after issuing chunk k+1 guarantees chunk k
// has landed in LDS. Manual staging fallback if the builtin is unavailable.
// MODE 0: f16 out, head-major rows  [(b*16+h)*256+n][dk]   (Q/K)
// MODE 1: f16 out, per-head transp. [(b*16+h)*64+dk][n]    (V)
// MODE 2: f32 out, plain [4096][1024] + bias               (final proj)
// ---------------------------------------------------------------------------
template <int MODE>
__global__ __launch_bounds__(256) void gemm_k(
    const _Float16* __restrict__ A, const _Float16* __restrict__ BT,
    const float* __restrict__ bias, void* __restrict__ outp)
{
  __shared__ __align__(16) _Float16 Ash[2][128][40];
  __shared__ __align__(16) _Float16 Bsh[2][128][40];
  const int tid = threadIdx.x;
  const int mblk = blockIdx.x * 128;
  const int nblk = blockIdx.y * 128;
  const int w  = tid >> 5, l = tid & 31;
  const int wm = (w & 3) * 32;       // 4 wave-rows of 32
  const int wn = (w >> 2) * 64;      // 2 wave-cols of 64
  const int lm = l & 15, lh = l >> 4;

  v8f acc[2][4] = {};

  const _Float16* Ab = A  + (size_t)mblk * 1024;
  const _Float16* Bb = BT + (size_t)nblk * 1024;

#if USE_TDM
  const unsigned ldsA0 = (unsigned)(uintptr_t)&Ash[0][0][0];
  const unsigned ldsA1 = (unsigned)(uintptr_t)&Ash[1][0][0];
  const unsigned ldsB0 = (unsigned)(uintptr_t)&Bsh[0][0][0];
  const unsigned ldsB1 = (unsigned)(uintptr_t)&Bsh[1][0][0];
  // Prologue: stream chunk 0 into buffer 0.
  if (w == 0) {
    tdm_load_tile_2d(ldsA0, Ab, 1024ull, 32u, 128u);
    tdm_load_tile_2d(ldsB0, Bb, 1024ull, 32u, 128u);
  }
#else
  const int c4   = (tid & 7) * 4;
  const int arow = tid >> 3;
#endif

  for (int kc = 0; kc < 32; ++kc) {
    const int buf = kc & 1;
#if USE_TDM
    if (w == 0) {
      if (kc < 31) {
        const int k1 = (kc + 1) * 32;
        tdm_load_tile_2d(buf ? ldsA0 : ldsA1, Ab + k1, 1024ull, 32u, 128u);
        tdm_load_tile_2d(buf ? ldsB0 : ldsB1, Bb + k1, 1024ull, 32u, 128u);
        __builtin_amdgcn_s_wait_tensorcnt(2);   // chunk kc complete
      } else {
        __builtin_amdgcn_s_wait_tensorcnt(0);   // last chunk complete
      }
    }
    __syncthreads();
#else
    {
      const int k0 = kc * 32;
      #pragma unroll
      for (int rr = 0; rr < 4; ++rr) {
        int r = arow + 32 * rr;
        *(v4h*)&Ash[buf][r][c4] = *(const v4h*)(Ab + (size_t)r * 1024 + k0 + c4);
        *(v4h*)&Bsh[buf][r][c4] = *(const v4h*)(Bb + (size_t)r * 1024 + k0 + c4);
      }
      __syncthreads();
    }
#endif

    v16h af[2], bf[4];
    #pragma unroll
    for (int tm = 0; tm < 2; ++tm) {
      const _Float16* ap = &Ash[buf][wm + tm * 16 + lm][lh * 8];
      af[tm] = cat16(*(const v8h*)ap, *(const v8h*)(ap + 16));
    }
    #pragma unroll
    for (int tn = 0; tn < 4; ++tn) {
      const _Float16* bp = &Bsh[buf][wn + tn * 16 + lm][lh * 8];
      bf[tn] = cat16(*(const v8h*)bp, *(const v8h*)(bp + 16));
    }
    #pragma unroll
    for (int tm = 0; tm < 2; ++tm)
      #pragma unroll
      for (int tn = 0; tn < 4; ++tn)
        acc[tm][tn] = wmma_f16(af[tm], bf[tn], acc[tm][tn]);
    __syncthreads();
  }

  #pragma unroll
  for (int tm = 0; tm < 2; ++tm)
    #pragma unroll
    for (int tn = 0; tn < 4; ++tn)
      #pragma unroll
      for (int r = 0; r < 8; ++r) {
        int gm = mblk + wm + tm * 16 + r + 8 * lh;   // row = b*256+n
        int gn = nblk + wn + tn * 16 + lm;           // col = h*64+dk
        float v = acc[tm][tn][r] + bias[gn];
        if (MODE == 2) {
          ((float*)outp)[(size_t)gm * 1024 + gn] = v;
        } else {
          int b = gm >> 8, nn = gm & 255, h = gn >> 6, dk = gn & 63;
          if (MODE == 0)
            ((_Float16*)outp)[(((size_t)(b * 16 + h) * 256) + nn) * 64 + dk] =
                (_Float16)v;
          else
            ((_Float16*)outp)[(((size_t)(b * 16 + h) * 64) + dk) * 256 + nn] =
                (_Float16)v;
        }
      }
}

// ---------------------------------------------------------------------------
// Fused attention per (b,h): S = QK^T/8 (masked), softmax in registers,
// P = LAM_ATT*softmax + pmix, X = P @ V.  One wave = 16 query rows.
// ---------------------------------------------------------------------------
__global__ __launch_bounds__(256) void attn_k(
    const _Float16* __restrict__ Qh, const _Float16* __restrict__ Kh,
    const _Float16* __restrict__ VhT, const _Float16* __restrict__ pmix,
    const int* __restrict__ mask, _Float16* __restrict__ Xh)
{
  __shared__ __align__(16) _Float16 Ps[8][16][40];  // per-wave P staging
  const int tid = threadIdx.x;
  const int w = tid >> 5, l = tid & 31;
  const int lm = l & 15, lh = l >> 4;
  const int bh = blockIdx.y;
  const int b = bh >> 4, h = bh & 15;
  const int m0 = blockIdx.x * 128 + w * 16;

  const _Float16* Qb = Qh  + (size_t)bh * (N_ * DK_);
  const _Float16* Kb = Kh  + (size_t)bh * (N_ * DK_);
  const _Float16* Vb = VhT + (size_t)bh * (DK_ * N_);

  v16h qf[2];
  {
    const _Float16* qp = Qb + (size_t)(m0 + lm) * DK_ + lh * 8;
    #pragma unroll
    for (int c = 0; c < 2; ++c)
      qf[c] = cat16(*(const v8h*)(qp + c * 32), *(const v8h*)(qp + c * 32 + 16));
  }

  v8f s[16];
  #pragma unroll
  for (int jn = 0; jn < 16; ++jn) {
    int n = jn * 16 + lm;
    v8f a = {};
    const _Float16* kp = Kb + (size_t)n * DK_ + lh * 8;
    #pragma unroll
    for (int c = 0; c < 2; ++c) {
      v16h bf = cat16(*(const v8h*)(kp + c * 32), *(const v8h*)(kp + c * 32 + 16));
      a = wmma_f16(qf[c], bf, a);
    }
    int mk = mask[b * N_ + n];
    #pragma unroll
    for (int r = 0; r < 8; ++r)
      s[jn][r] = mk ? a[r] * 0.125f : -1e12f;
  }

  float mx[8], sc[8];
  #pragma unroll
  for (int r = 0; r < 8; ++r) {
    float m_ = -3.0e38f;
    #pragma unroll
    for (int jn = 0; jn < 16; ++jn) m_ = fmaxf(m_, s[jn][r]);
    #pragma unroll
    for (int off = 8; off >= 1; off >>= 1) m_ = fmaxf(m_, __shfl_xor(m_, off, 32));
    float su = 0.f;
    #pragma unroll
    for (int jn = 0; jn < 16; ++jn) su += __expf(s[jn][r] - m_);
    #pragma unroll
    for (int off = 8; off >= 1; off >>= 1) su += __shfl_xor(su, off, 32);
    mx[r] = m_;
    sc[r] = LAM_ATT / su;
  }

  v8f xacc[4] = {};
  for (int kc = 0; kc < 8; ++kc) {
    #pragma unroll
    for (int t = 0; t < 2; ++t) {
      int jn = kc * 2 + t;
      int n = jn * 16 + lm;
      #pragma unroll
      for (int r = 0; r < 8; ++r) {
        int mrow = m0 + r + 8 * lh;
        float pm = (float)pmix[((size_t)b * N_ + mrow) * N_ + n];
        float p = __expf(s[jn][r] - mx[r]) * sc[r] + pm;
        Ps[w][r + 8 * lh][t * 16 + lm] = (_Float16)p;
      }
    }
    __syncthreads();
    const _Float16* pp = &Ps[w][lm][lh * 8];
    v16h pf = cat16(*(const v8h*)pp, *(const v8h*)(pp + 16));
    #pragma unroll
    for (int jd = 0; jd < 4; ++jd) {
      int dk = jd * 16 + lm;
      const _Float16* vp = Vb + (size_t)dk * N_ + kc * 32 + lh * 8;
      v16h vf = cat16(*(const v8h*)vp, *(const v8h*)(vp + 16));
      xacc[jd] = wmma_f16(pf, vf, xacc[jd]);
    }
    __syncthreads();
  }

  #pragma unroll
  for (int jd = 0; jd < 4; ++jd)
    #pragma unroll
    for (int r = 0; r < 8; ++r) {
      int row = b * N_ + m0 + r + 8 * lh;
      int col = h * DK_ + jd * 16 + lm;
      Xh[(size_t)row * D_ + col] = (_Float16)xacc[jd][r];
    }
}

// ---------------------------------------------------------------------------
extern "C" void kernel_launch(void* const* d_in, const int* in_sizes, int n_in,
                              void* d_out, int out_size, void* d_ws, size_t ws_size,
                              hipStream_t stream)
{
  const float* query = (const float*)d_in[0];
  const float* key   = (const float*)d_in[1];
  const float* value = (const float*)d_in[2];
  const float* adj   = (const float*)d_in[3];
  const float* dist  = (const float*)d_in[4];
  const int*   mask  = (const int*)d_in[6];
  const float* Wq = (const float*)d_in[7];
  const float* bq = (const float*)d_in[8];
  const float* Wk = (const float*)d_in[9];
  const float* bk = (const float*)d_in[10];
  const float* Wv = (const float*)d_in[11];
  const float* bv = (const float*)d_in[12];
  const float* Wo = (const float*)d_in[13];
  const float* bo = (const float*)d_in[14];

  const size_t XSZ = (size_t)B_ * N_ * D_;       // 4,194,304 halves
  const size_t WSZ = (size_t)D_ * D_;            // 1,048,576 halves
  _Float16* Xq   = (_Float16*)d_ws;              // also reused as Xh
  _Float16* Xk   = Xq  + XSZ;
  _Float16* Xv   = Xk  + XSZ;
  _Float16* WqT  = Xv  + XSZ;
  _Float16* WkT  = WqT + WSZ;
  _Float16* WvT  = WkT + WSZ;
  _Float16* WoT  = WvT + WSZ;
  _Float16* Qh   = WoT + WSZ;
  _Float16* Kh   = Qh  + XSZ;
  _Float16* VhT  = Kh  + XSZ;
  _Float16* pmix = VhT + XSZ;                    // 1,048,576 halves
  _Float16* Xh   = Xq;                           // Xq is dead after gemm<0>

  cvt_f32_f16_k<<<dim3(4096), 256, 0, stream>>>(query, Xq);
  cvt_f32_f16_k<<<dim3(4096), 256, 0, stream>>>(key,   Xk);
  cvt_f32_f16_k<<<dim3(4096), 256, 0, stream>>>(value, Xv);
  cvt_w_tr_k<<<dim3(32, 32), 256, 0, stream>>>(Wq, WqT);
  cvt_w_tr_k<<<dim3(32, 32), 256, 0, stream>>>(Wk, WkT);
  cvt_w_tr_k<<<dim3(32, 32), 256, 0, stream>>>(Wv, WvT);
  cvt_w_tr_k<<<dim3(32, 32), 256, 0, stream>>>(Wo, WoT);

  gemm_k<0><<<dim3(32, 8), 256, 0, stream>>>(Xq, WqT, bq, Qh);
  gemm_k<0><<<dim3(32, 8), 256, 0, stream>>>(Xk, WkT, bk, Kh);
  gemm_k<1><<<dim3(32, 8), 256, 0, stream>>>(Xv, WvT, bv, VhT);

  pmix_k<<<dim3(B_ * N_ / 8), 256, 0, stream>>>(adj, dist, mask, pmix);
  attn_k<<<dim3(2, 256), 256, 0, stream>>>(Qh, Kh, VhT, pmix, mask, Xh);

  gemm_k<2><<<dim3(32, 8), 256, 0, stream>>>(Xh, WoT, bo, (float*)d_out);
}
